// ObjectOrientedHierarchicalVAE_14645838479609
// MI455X (gfx1250) — matrix operand order repaired and checked by reading, not compiled, not run on hardware
//
#include <hip/hip_runtime.h>
#include <hip/hip_bf16.h>

typedef _Float16 v8h  __attribute__((ext_vector_type(8)));
typedef _Float16 v16h __attribute__((ext_vector_type(16)));
typedef float    v8f  __attribute__((ext_vector_type(8)));

#define MOBJ     20
#define NTHREADS 256
#define NWAVES   8
#define NMT      25     /* 400 pairs = 25 tiles of 16 */

/* ---- dynamic LDS layout (bytes) ---- */
#define OFF_WGT   0                    /* 32768: fcw^T -> W1_top^T -> W1_bot^T -> W2^T (f16, [n][k]) */
#define OFF_U     32768                /* 32*128*4 = 16384 (rows 20..31 dead) */
#define OFF_V     (OFF_U + 16384)      /* 16384 */
#define OFF_SCR   (OFF_V + 16384)      /* 65536: temps, then 8x8KB pair-stage wave scratch */
#define OFF_RSUM  (OFF_SCR + 65536)    /* 128 f32 */
#define OFF_STATS (OFF_RSUM + 512)     /* 64 f32 */
#define OFF_MASK  (OFF_STATS + 256)    /* 20 f32 (padded) */
#define LDS_BYTES (OFF_MASK + 128)     /* = 131968 -> 2 WGs / WGP */

/* temporaries inside the scratch region (pre-pair phases only) */
#define SOFF_X     0        /* x      [20][128] f32 = 10240 */
#define SOFF_PRELN 10240    /* pre-LN [32][128] f32 = 16384 (rows 20..31 dead) */
#define SOFF_H16   26624    /* xh/featsh [32][128] f16 = 8192 (rows 20..31 zero) */
#define SOFF_TMPA  34816    /* [20][64] f32 = 5120 */
#define SOFF_TMPB  39936    /* [20][32] f32 = 2560 */

struct P {
  const float* bbox; const float* objsz; const int* color; const int* omask;
  const float *oew1,*oeb1,*oeg1,*oebb1,*oew2,*oeb2,*oeg2,*oebb2;
  const float *ce;
  const float *sew,*seb,*seg,*sebb;
  const float *fcw,*fcb,*fcg,*fcbb;
  const float *rew1,*reb1,*reg1,*rebb1,*rew2,*reb2,*reg2,*rebb2;
  const float *gaw,*gab,*gag,*gabb;
  float* out;
};

/* A fragment (16-bit A layout): lane half -> K 0-7/16-23 vs 8-15/24-31 */
static __device__ inline v16h ld_afrag(const _Float16* arow, int kc, int halfsel) {
  int kb = kc * 32 + halfsel * 8;
  v8h lo = *(const v8h*)(arow + kb);
  v8h hi = *(const v8h*)(arow + kb + 16);
  return __builtin_shufflevector(lo, hi, 0, 1, 2, 3, 4, 5, 6, 7,
                                 8, 9, 10, 11, 12, 13, 14, 15);
}
/* B fragment (16-bit B layout): lane half -> K 0-7/16-23 vs 8-15/24-31 */
static __device__ inline v16h ld_bfrag(const _Float16* wrow, int kc, int halfsel) {
  int kb = kc * 32 + halfsel * 8;
  v8h lo = *(const v8h*)(wrow + kb);
  v8h hi = *(const v8h*)(wrow + kb + 16);
  return __builtin_shufflevector(lo, hi, 0, 1, 2, 3, 4, 5, 6, 7,
                                 8, 9, 10, 11, 12, 13, 14, 15);
}
/* one 16x16 output tile of [32,128] x WT[128,128] (K = 128): preload all
   fragments, then a 4-deep wmma chain -> one ds wait per tile, good ILP. */
static __device__ inline v8f wmma_tile(const _Float16* Ah, const _Float16* WT,
                                       int mtile, int ntile, int lane) {
  const int r = lane & 15, halfsel = lane >> 4;
  const _Float16* arow = Ah + (mtile * 16 + r) * 128;
  const _Float16* wrow = WT + (ntile * 16 + r) * 128;
  v16h a[4], bf[4];
#pragma unroll
  for (int kc = 0; kc < 4; ++kc) {
    a[kc]  = ld_afrag(arow, kc, halfsel);
    bf[kc] = ld_bfrag(wrow, kc, halfsel);
  }
  v8f cc = {0.f, 0.f, 0.f, 0.f, 0.f, 0.f, 0.f, 0.f};
#pragma unroll
  for (int kc = 0; kc < 4; ++kc)
    cc = __builtin_amdgcn_wmma_f32_16x16x32_f16(false, a[kc], false, bf[kc],
                                                (short)0, cc, false, false);
  return cc;
}

__global__ __launch_bounds__(NTHREADS)
void relnet_kernel(P p) {
  extern __shared__ char smem[];
  const int b    = blockIdx.x;
  const int tid  = threadIdx.x;
  const int lane = tid & 31;
  const int wave = tid >> 5;

  _Float16* wgt    = (_Float16*)(smem + OFF_WGT);
  float*    Ub     = (float*)(smem + OFF_U);
  float*    Vb     = (float*)(smem + OFF_V);
  float*    relsum = (float*)(smem + OFF_RSUM);
  float*    stats  = (float*)(smem + OFF_STATS);
  float*    maskf  = (float*)(smem + OFF_MASK);
  float*    xb     = (float*)(smem + OFF_SCR + SOFF_X);
  float*    preln  = (float*)(smem + OFF_SCR + SOFF_PRELN);
  _Float16* h16    = (_Float16*)(smem + OFF_SCR + SOFF_H16);
  float*    tmpA   = (float*)(smem + OFF_SCR + SOFF_TMPA);
  float*    tmpB   = (float*)(smem + OFF_SCR + SOFF_TMPB);

  /* warm L2 for the hot weights (global_prefetch_b8) */
  __builtin_prefetch(p.rew2 + (tid << 6), 0, 1);
  __builtin_prefetch(p.rew1 + (tid << 7), 0, 1);
  __builtin_prefetch(p.fcw  + (tid << 6), 0, 1);

  /* ---- phase 0: stage fcw^T (f16) + masks + zero rel_sum ---- */
  for (int idx = tid; idx < 128 * 128; idx += NTHREADS) {
    int n = idx >> 7, k = idx & 127;
    wgt[n * 128 + k] = (_Float16)p.fcw[k * 128 + n];
  }
  if (tid < MOBJ) maskf[tid] = (float)p.omask[b * MOBJ + tid];
  if (tid < 128)  relsum[tid] = 0.f;
  __syncthreads();
  if (tid == 0) {
    float S = 0.f;
    for (int o = 0; o < MOBJ; ++o) S += maskf[o];
    stats[40] = 0.5f * S * (S - 1.f);   /* #valid upper-tri pairs */
  }

  /* ---- object encoder: bbox 4->32, LN, ReLU -> tmpB ---- */
  for (int idx = tid; idx < MOBJ * 32; idx += NTHREADS) {
    int o = idx >> 5, c = idx & 31;
    const float* bbp = p.bbox + (b * MOBJ + o) * 4;
    float acc = p.oeb1[c];
    for (int k = 0; k < 4; ++k) acc += bbp[k] * p.oew1[k * 32 + c];
    tmpA[o * 32 + c] = acc;
  }
  __syncthreads();
  if (tid < MOBJ) {
    float s = 0.f, s2 = 0.f;
    for (int c = 0; c < 32; ++c) { float v = tmpA[tid * 32 + c]; s += v; s2 += v * v; }
    float mu = s * (1.f / 32.f), var = s2 * (1.f / 32.f) - mu * mu;
    stats[tid] = mu; stats[20 + tid] = rsqrtf(var + 1e-5f);
  }
  __syncthreads();
  for (int idx = tid; idx < MOBJ * 32; idx += NTHREADS) {
    int o = idx >> 5, c = idx & 31;
    float v = (tmpA[o * 32 + c] - stats[o]) * stats[20 + o] * p.oeg1[c] + p.oebb1[c];
    tmpB[o * 32 + c] = v > 0.f ? v : 0.f;
  }
  __syncthreads();

  /* ---- 32->64, LN, ReLU -> xb[:, 0:64] ---- */
  for (int idx = tid; idx < MOBJ * 64; idx += NTHREADS) {
    int o = idx >> 6, c = idx & 63;
    float acc = p.oeb2[c];
    for (int k = 0; k < 32; ++k) acc += tmpB[o * 32 + k] * p.oew2[k * 64 + c];
    tmpA[o * 64 + c] = acc;
  }
  __syncthreads();
  if (tid < MOBJ) {
    float s = 0.f, s2 = 0.f;
    for (int c = 0; c < 64; ++c) { float v = tmpA[tid * 64 + c]; s += v; s2 += v * v; }
    float mu = s * (1.f / 64.f), var = s2 * (1.f / 64.f) - mu * mu;
    stats[tid] = mu; stats[20 + tid] = rsqrtf(var + 1e-5f);
  }
  __syncthreads();
  for (int idx = tid; idx < MOBJ * 64; idx += NTHREADS) {
    int o = idx >> 6, c = idx & 63;
    float v = (tmpA[o * 64 + c] - stats[o]) * stats[20 + o] * p.oeg2[c] + p.oebb2[c];
    xb[o * 128 + c] = v > 0.f ? v : 0.f;
  }
  __syncthreads();

  /* ---- color embedding -> xb[:, 64:96]; size pre-LN -> tmpA ---- */
  for (int idx = tid; idx < MOBJ * 32; idx += NTHREADS) {
    int o = idx >> 5, c = idx & 31;
    xb[o * 128 + 64 + c] = p.ce[p.color[b * MOBJ + o] * 32 + c];
    tmpA[o * 32 + c] = p.objsz[b * MOBJ + o] * p.sew[c] + p.seb[c];
  }
  __syncthreads();
  if (tid < MOBJ) {
    float s = 0.f, s2 = 0.f;
    for (int c = 0; c < 32; ++c) { float v = tmpA[tid * 32 + c]; s += v; s2 += v * v; }
    float mu = s * (1.f / 32.f), var = s2 * (1.f / 32.f) - mu * mu;
    stats[tid] = mu; stats[20 + tid] = rsqrtf(var + 1e-5f);
  }
  __syncthreads();
  for (int idx = tid; idx < MOBJ * 32; idx += NTHREADS) {
    int o = idx >> 5, c = idx & 31;
    float v = (tmpA[o * 32 + c] - stats[o]) * stats[20 + o] * p.seg[c] + p.sebb[c];
    xb[o * 128 + 96 + c] = v > 0.f ? v : 0.f;
  }
  __syncthreads();

  /* ---- x -> f16 tile (rows 20..31 zero) ---- */
  for (int idx = tid; idx < 32 * 128; idx += NTHREADS) {
    int row = idx >> 7, c = idx & 127;
    h16[idx] = (row < MOBJ) ? (_Float16)xb[row * 128 + c] : (_Float16)0.f;
  }
  __syncthreads();

  /* ---- feature combiner 128->128 on WMMA: wave owns ntile == wave ---- */
  {
    const int n = wave * 16 + (lane & 15);
    const float bias = p.fcb[n];
#pragma unroll
    for (int mtile = 0; mtile < 2; ++mtile) {
      v8f cc = wmma_tile(h16, wgt, mtile, wave, lane);
#pragma unroll
      for (int rv = 0; rv < 8; ++rv) {
        int row = mtile * 16 + rv + (lane >> 4) * 8;
        preln[row * 128 + n] = cc[rv] + bias;   /* rows 20..31 dead, no guard */
      }
    }
  }
  __syncthreads();

  /* LN stats for fc; concurrently stage W1_top^T over fcw^T */
  if (tid < MOBJ) {
    float s = 0.f, s2 = 0.f;
    for (int c = 0; c < 128; ++c) { float v = preln[tid * 128 + c]; s += v; s2 += v * v; }
    float mu = s * (1.f / 128.f), var = s2 * (1.f / 128.f) - mu * mu;
    stats[tid] = mu; stats[20 + tid] = rsqrtf(var + 1e-5f);
  }
  for (int idx = tid; idx < 128 * 128; idx += NTHREADS) {
    int n = idx >> 7, k = idx & 127;
    wgt[n * 128 + k] = (_Float16)p.rew1[k * 128 + n];
  }
  __syncthreads();
  /* normalize + ReLU -> featsh (f16, reuses h16; rows 20..31 still zero) */
  for (int idx = tid; idx < MOBJ * 128; idx += NTHREADS) {
    int o = idx >> 7, c = idx & 127;
    float v = (preln[o * 128 + c] - stats[o]) * stats[20 + o] * p.fcg[c] + p.fcbb[c];
    h16[o * 128 + c] = (_Float16)(v > 0.f ? v : 0.f);
  }
  __syncthreads();

  /* ---- U = feats@W1_top + b1 on WMMA ---- */
  {
    const int n = wave * 16 + (lane & 15);
    const float bias = p.reb1[n];
#pragma unroll
    for (int mtile = 0; mtile < 2; ++mtile) {
      v8f cc = wmma_tile(h16, wgt, mtile, wave, lane);
#pragma unroll
      for (int rv = 0; rv < 8; ++rv) {
        int row = mtile * 16 + rv + (lane >> 4) * 8;
        Ub[row * 128 + n] = cc[rv] + bias;
      }
    }
  }
  __syncthreads();
  /* stage W1_bot^T */
  for (int idx = tid; idx < 128 * 128; idx += NTHREADS) {
    int n = idx >> 7, k = idx & 127;
    wgt[n * 128 + k] = (_Float16)p.rew1[(128 + k) * 128 + n];
  }
  __syncthreads();
  /* ---- V = feats@W1_bot on WMMA ---- */
  {
    const int n = wave * 16 + (lane & 15);
#pragma unroll
    for (int mtile = 0; mtile < 2; ++mtile) {
      v8f cc = wmma_tile(h16, wgt, mtile, wave, lane);
#pragma unroll
      for (int rv = 0; rv < 8; ++rv) {
        int row = mtile * 16 + rv + (lane >> 4) * 8;
        Vb[row * 128 + n] = cc[rv];
      }
    }
  }
  __syncthreads();
  /* ---- stage re_w2^T over the weight region ---- */
  for (int idx = tid; idx < 128 * 128; idx += NTHREADS) {
    int n = idx >> 7, k = idx & 127;
    wgt[n * 128 + k] = (_Float16)p.rew2[k * 128 + n];
  }
  __syncthreads();

  /* ---- pair stage: per-wave 16-pair tiles, WMMA for rel1@W2 ---- */
  float accq[4] = {0.f, 0.f, 0.f, 0.f};   /* lane owns cols lane+32q */
  {
    float*    scrf = (float*)(smem + OFF_SCR + wave * 8192);
    _Float16* scrh = (_Float16*)(smem + OFF_SCR + wave * 8192);
    const int r       = lane & 15;
    const int halfsel = lane >> 4;
    const int cbase   = halfsel * 64;
    const int ncol    = lane & 15;

    for (int mt = wave; mt < NMT; mt += NWAVES) {
      int pp = mt * 16 + r;
      int i  = pp / MOBJ;
      int j  = pp - i * MOBJ;
      const float* Ui = Ub + i * 128;
      const float* Vj = Vb + j * 128;

      /* LN(U[i]+V[j]) + ReLU -> f16 tile (2 lanes per row) */
      float s = 0.f, s2 = 0.f;
      for (int c = cbase; c < cbase + 64; ++c) {
        float v = Ui[c] + Vj[c]; s += v; s2 += v * v;
      }
      s  += __shfl_xor(s, 16);
      s2 += __shfl_xor(s2, 16);
      float mu  = s * (1.f / 128.f);
      float var = s2 * (1.f / 128.f) - mu * mu;
      float rs  = rsqrtf(var + 1e-5f);
      for (int c = cbase; c < cbase + 64; ++c) {
        float v = (Ui[c] + Vj[c] - mu) * rs * p.reg1[c] + p.rebb1[c];
        scrh[r * 128 + c] = (_Float16)(v > 0.f ? v : 0.f);
      }
      asm volatile("s_wait_dscnt 0x0" ::: "memory");

      v16h afrag[4];
#pragma unroll
      for (int kc = 0; kc < 4; ++kc)
        afrag[kc] = ld_afrag(scrh + r * 128, kc, halfsel);

      v8f acc[8];
#pragma unroll
      for (int nt = 0; nt < 8; ++nt) {
        v8f cc = {0.f, 0.f, 0.f, 0.f, 0.f, 0.f, 0.f, 0.f};
        const _Float16* wrow = wgt + (nt * 16 + ncol) * 128;
#pragma unroll
        for (int kc = 0; kc < 4; ++kc) {
          v16h bfrag = ld_bfrag(wrow, kc, halfsel);
          cc = __builtin_amdgcn_wmma_f32_16x16x32_f16(
                 false, afrag[kc], false, bfrag, (short)0, cc, false, false);
        }
        acc[nt] = cc;
      }

      /* h2 + bias -> f32 tile (overwrites rel1; A frags live in regs) */
#pragma unroll
      for (int nt = 0; nt < 8; ++nt) {
        int   n    = nt * 16 + ncol;
        float bias = p.reb2[n];
#pragma unroll
        for (int rv = 0; rv < 8; ++rv) {
          int row = rv + halfsel * 8;
          scrf[row * 128 + n] = acc[nt][rv] + bias;
        }
      }
      asm volatile("s_wait_dscnt 0x0" ::: "memory");

      /* LN + ReLU in place */
      s = 0.f; s2 = 0.f;
      for (int c = cbase; c < cbase + 64; ++c) {
        float v = scrf[r * 128 + c]; s += v; s2 += v * v;
      }
      s  += __shfl_xor(s, 16);
      s2 += __shfl_xor(s2, 16);
      mu  = s * (1.f / 128.f);
      var = s2 * (1.f / 128.f) - mu * mu;
      rs  = rsqrtf(var + 1e-5f);
      for (int c = cbase; c < cbase + 64; ++c) {
        float v = (scrf[r * 128 + c] - mu) * rs * p.reg2[c] + p.rebb2[c];
        scrf[r * 128 + c] = v > 0.f ? v : 0.f;
      }
      asm volatile("s_wait_dscnt 0x0" ::: "memory");

      /* masked (upper-triangle) column accumulation */
      for (int rr = 0; rr < 16; ++rr) {
        int   q2 = mt * 16 + rr;
        int   ii = q2 / MOBJ;
        int   jj = q2 - ii * MOBJ;
        float w  = (jj > ii) ? maskf[ii] * maskf[jj] : 0.f;
        if (w != 0.f) {
#pragma unroll
          for (int q = 0; q < 4; ++q)
            accq[q] += w * scrf[rr * 128 + q * 32 + lane];
        }
      }
    }
  }
#pragma unroll
  for (int q = 0; q < 4; ++q) atomicAdd(&relsum[q * 32 + lane], accq[q]);
  __syncthreads();

  /* ---- global aggregator 128->128, LN, ReLU, cnt gate ---- */
  float  cnt  = stats[40];
  float* ybuf = (float*)(smem + OFF_SCR);   /* scratch is free now */
  if (tid < 128) {
    float inv = 1.f / fmaxf(cnt, 1.f);
    float acc = p.gab[tid];
    for (int k = 0; k < 128; ++k) acc += (relsum[k] * inv) * p.gaw[k * 128 + tid];
    ybuf[tid] = acc;
  }
  __syncthreads();
  if (tid == 0) {
    float s = 0.f, s2 = 0.f;
    for (int c = 0; c < 128; ++c) { float v = ybuf[c]; s += v; s2 += v * v; }
    float mu = s * (1.f / 128.f), var = s2 * (1.f / 128.f) - mu * mu;
    stats[41] = mu; stats[42] = rsqrtf(var + 1e-5f);
  }
  __syncthreads();
  if (tid < 128) {
    float v = (ybuf[tid] - stats[41]) * stats[42] * p.gag[tid] + p.gabb[tid];
    v = v > 0.f ? v : 0.f;
    p.out[b * 128 + tid] = (cnt > 0.f) ? v : 0.f;
  }
}

extern "C" void kernel_launch(void* const* d_in, const int* in_sizes, int n_in,
                              void* d_out, int out_size, void* d_ws, size_t ws_size,
                              hipStream_t stream) {
  (void)n_in; (void)out_size; (void)d_ws; (void)ws_size;
  P p;
  p.bbox  = (const float*)d_in[0];
  p.objsz = (const float*)d_in[1];
  p.color = (const int*)d_in[2];
  p.omask = (const int*)d_in[3];
  p.oew1  = (const float*)d_in[4];  p.oeb1  = (const float*)d_in[5];
  p.oeg1  = (const float*)d_in[6];  p.oebb1 = (const float*)d_in[7];
  p.oew2  = (const float*)d_in[8];  p.oeb2  = (const float*)d_in[9];
  p.oeg2  = (const float*)d_in[10]; p.oebb2 = (const float*)d_in[11];
  p.ce    = (const float*)d_in[12];
  p.sew   = (const float*)d_in[13]; p.seb   = (const float*)d_in[14];
  p.seg   = (const float*)d_in[15]; p.sebb  = (const float*)d_in[16];
  p.fcw   = (const float*)d_in[17]; p.fcb   = (const float*)d_in[18];
  p.fcg   = (const float*)d_in[19]; p.fcbb  = (const float*)d_in[20];
  p.rew1  = (const float*)d_in[21]; p.reb1  = (const float*)d_in[22];
  p.reg1  = (const float*)d_in[23]; p.rebb1 = (const float*)d_in[24];
  p.rew2  = (const float*)d_in[25]; p.reb2  = (const float*)d_in[26];
  p.reg2  = (const float*)d_in[27]; p.rebb2 = (const float*)d_in[28];
  p.gaw   = (const float*)d_in[29]; p.gab   = (const float*)d_in[30];
  p.gag   = (const float*)d_in[31]; p.gabb  = (const float*)d_in[32];
  p.out   = (float*)d_out;

  const int Bn = in_sizes[2] / MOBJ;   /* color is [B, M] */
  hipLaunchKernelGGL(relnet_kernel, dim3(Bn), dim3(NTHREADS), LDS_BYTES, stream, p);
}
